// SemiSeparableTokenMix_43748536877634
// MI455X (gfx1250) — compile-verified, matrix operand order checked by reading, not run on hardware
//
#include <hip/hip_runtime.h>

typedef __attribute__((ext_vector_type(16))) __bf16 v16bf;
typedef __attribute__((ext_vector_type(8)))  __bf16 v8bf;
typedef __attribute__((ext_vector_type(8)))  float  v8f;
typedef __attribute__((ext_vector_type(4)))  unsigned int u32x4;
typedef __attribute__((ext_vector_type(8)))  int i32x8;
typedef __attribute__((ext_vector_type(4)))  int i32x4;

#define D_DIM   512
#define B_SZ    4
#define T_LEN   8192
#define M_DIM   (B_SZ * T_LEN)   // 32768
#define CHUNK   128
#define NCHUNK  (T_LEN / CHUNK)  // 64
#define TM      256              // block M tile (8 waves x 32 rows)
#define TN      64               // block N tile (4 x 16 cols per wave)
#define LDSPAD  8                // bf16 elements of row padding (16B) to spread banks

// ---------------------------------------------------------------------------
// Conversions / precompute
// ---------------------------------------------------------------------------
__global__ void cvt_f32_to_bf16(const float* __restrict__ src,
                                __bf16* __restrict__ dst, int n) {
    int i = blockIdx.x * blockDim.x + threadIdx.x;
    if (i < n) dst[i] = (__bf16)src[i];
}

__global__ void prep_decay(const float* __restrict__ A,
                           float* __restrict__ a_sig,
                           float* __restrict__ a_pow) {
    int d = blockIdx.x * blockDim.x + threadIdx.x;
    if (d < D_DIM) {
        float s = 1.0f / (1.0f + __expf(-A[d]));
        a_sig[d] = s;
        a_pow[d] = __powf(s, (float)CHUNK);
    }
}

// ---------------------------------------------------------------------------
// TDM: 2D tile load Global -> LDS (bf16 elements), with per-row LDS padding.
// Descriptor bitfields per CDNA5 ISA.
// ---------------------------------------------------------------------------
__device__ __forceinline__ void tdm_load_2d_bf16(unsigned lds_off,
                                                 const void* gptr,
                                                 unsigned tile_d0,   // elems per row
                                                 unsigned tile_d1,   // rows
                                                 unsigned tensor_d0,
                                                 unsigned tensor_d1,
                                                 unsigned stride0,   // elems
                                                 unsigned pad_interval, // 7 -> 256 DW
                                                 unsigned pad_amount) { // 3 -> 4 DW
    unsigned long long ga = (unsigned long long)gptr;
    u32x4 g0;
    g0[0] = 1u;                                   // count=1, user descriptor
    g0[1] = lds_off;                              // lds_addr
    g0[2] = (unsigned)(ga & 0xffffffffu);         // global_addr[31:0]
    g0[3] = (unsigned)((ga >> 32) & 0x01ffffffu)  // global_addr[56:32]
          | (2u << 30);                           // type = 2 ("image")
    i32x8 g1;
    g1[0] = (int)((1u << 16)                      // data_size = 1 -> 2 bytes
          | (1u << 20)                            // pad_enable
          | ((pad_interval & 7u) << 22)
          | ((pad_amount & 127u) << 25));
    g1[1] = (int)((tensor_d0 & 0xffffu) << 16);   // [63:48] = tensor_dim0 lo16
    g1[2] = (int)(((tensor_d0 >> 16) & 0xffffu) | ((tensor_d1 & 0xffffu) << 16));
    g1[3] = (int)(((tensor_d1 >> 16) & 0xffffu) | ((tile_d0 & 0xffffu) << 16));
    g1[4] = (int)(tile_d1 & 0xffffu);             // tile_dim1 (tile_dim2 = 0)
    g1[5] = (int)stride0;                         // tensor_dim0_stride[31:0]
    g1[6] = 0;
    g1[7] = 0;
    i32x4 z4 = {0, 0, 0, 0};
    i32x8 z8 = {0, 0, 0, 0, 0, 0, 0, 0};
    __builtin_amdgcn_tensor_load_to_lds(g0, g1, z4, z4, z8, 0);
}

// ---------------------------------------------------------------------------
// WMMA GEMM:  C[M x 512] = A[M x 512] (bf16) * W[512 x 512]^T (bf16), fp32 out
// Block: 256 threads = 8 waves. Block tile 256(M) x 64(N).
// Wave computes 32 x 64: two A fragments share every B fragment -> 8 WMMA per
// K-step against 8 LDS b128 loads. W strip staged in LDS once via TDM.
// ---------------------------------------------------------------------------
__global__ __launch_bounds__(256)
void gemm_bf16_wmma(const __bf16* __restrict__ Amat,
                    const __bf16* __restrict__ Wmat,
                    float* __restrict__ Cmat) {
    __shared__ __bf16 sW[TN][D_DIM + LDSPAD];

    const int tid  = threadIdx.x;
    const int wave = tid >> 5;
    const int lane = tid & 31;
    const int n0   = blockIdx.x * TN;
    const int m0   = blockIdx.y * TM + wave * 32;

    // One wave DMAs the whole W strip (rows n0..n0+63, all K) into padded LDS.
    if (wave == 0) {
        unsigned lds_off = (unsigned)(unsigned long long)(uintptr_t)(void*)&sW[0][0];
        tdm_load_2d_bf16(lds_off, Wmat + (size_t)n0 * D_DIM,
                         /*tile_d0=*/D_DIM, /*tile_d1=*/TN,
                         /*tensor_d0=*/D_DIM, /*tensor_d1=*/D_DIM,
                         /*stride0=*/D_DIM,
                         /*pad_interval=*/7 /*256 DW = 1024B row*/,
                         /*pad_amount=*/3  /*4 DW = 16B = LDSPAD*/);
        __builtin_amdgcn_s_wait_tensorcnt(0);
    }
    __syncthreads();

    v8f acc[2][4];
#pragma unroll
    for (int mh = 0; mh < 2; ++mh)
#pragma unroll
        for (int i = 0; i < 4; ++i)
#pragma unroll
            for (int j = 0; j < 8; ++j) acc[mh][i][j] = 0.0f;

    // ISA 16-bit A/B fragment layout (wave32):
    //  lane < 16 : row (lane),    K chunks {k0..k0+7} and {k0+16..k0+23}
    //  lane >=16 : row (lane-16), K chunks {k0+8..k0+15} and {k0+24..k0+31}
    const int half = (lane >> 4) & 1;
    const int lrow = lane & 15;
    const __bf16* arow0 = Amat + (size_t)(m0 + lrow) * D_DIM;
    const __bf16* arow1 = arow0 + 16 * D_DIM;

    for (int k0 = 0; k0 < D_DIM; k0 += 32) {
        v16bf af0, af1;
        {
            const v8bf a0 = *reinterpret_cast<const v8bf*>(arow0 + k0 + half * 8);
            const v8bf a1 = *reinterpret_cast<const v8bf*>(arow0 + k0 + 16 + half * 8);
            const v8bf a2 = *reinterpret_cast<const v8bf*>(arow1 + k0 + half * 8);
            const v8bf a3 = *reinterpret_cast<const v8bf*>(arow1 + k0 + 16 + half * 8);
#pragma unroll
            for (int i = 0; i < 8; ++i) {
                af0[i] = a0[i]; af0[i + 8] = a1[i];
                af1[i] = a2[i]; af1[i + 8] = a3[i];
            }
        }
#pragma unroll
        for (int nt = 0; nt < 4; ++nt) {
            const __bf16* wrow = &sW[nt * 16 + lrow][0];
            const v8bf b0 = *reinterpret_cast<const v8bf*>(wrow + k0 + half * 8);
            const v8bf b1 = *reinterpret_cast<const v8bf*>(wrow + k0 + 16 + half * 8);
            v16bf bfv;
#pragma unroll
            for (int i = 0; i < 8; ++i) { bfv[i] = b0[i]; bfv[i + 8] = b1[i]; }
            acc[0][nt] = __builtin_amdgcn_wmma_f32_16x16x32_bf16(
                false, af0, false, bfv, (short)0, acc[0][nt], false, false);
            acc[1][nt] = __builtin_amdgcn_wmma_f32_16x16x32_bf16(
                false, af1, false, bfv, (short)0, acc[1][nt], false, false);
        }
    }

    // C/D layout: VGPR i holds M = i (lanes 0-15) / M = 8+i (lanes 16-31), N = lane&15
#pragma unroll
    for (int mh = 0; mh < 2; ++mh) {
        const int rbase = m0 + mh * 16 + half * 8;
#pragma unroll
        for (int nt = 0; nt < 4; ++nt) {
            const int col = n0 + nt * 16 + lrow;
#pragma unroll
            for (int i = 0; i < 8; ++i)
                Cmat[(size_t)(rbase + i) * D_DIM + col] = acc[mh][nt][i];
        }
    }
}

// ---------------------------------------------------------------------------
// Chunked diagonal scan: h_t = a*h_{t-1} + b_t  (per batch, per channel)
// ---------------------------------------------------------------------------
__global__ void scan_partial(const float* __restrict__ b,
                             const float* __restrict__ a_sig,
                             float* __restrict__ partial) {
    const int gid  = blockIdx.x * blockDim.x + threadIdx.x; // B*NCHUNK*D threads
    const int d    = gid % D_DIM;
    const int rest = gid / D_DIM;
    const int c    = rest % NCHUNK;
    const int bb   = rest / NCHUNK;
    const float av = a_sig[d];
    const float* p = b + ((bb * T_LEN + c * CHUNK) * D_DIM + d);
    float h = 0.0f;
    for (int t = 0; t < CHUNK; ++t) h = fmaf(av, h, p[t * D_DIM]);
    partial[(bb * NCHUNK + c) * D_DIM + d] = h;
}

__global__ void scan_carry(const float* __restrict__ partial,
                           const float* __restrict__ a_pow,
                           float* __restrict__ carry) {
    const int gid = blockIdx.x * blockDim.x + threadIdx.x; // B*D threads
    const int d   = gid % D_DIM;
    const int bb  = gid / D_DIM;
    const float ap = a_pow[d];
    float cv = 0.0f;
    for (int c = 0; c < NCHUNK; ++c) {
        carry[(bb * NCHUNK + c) * D_DIM + d] = cv;
        cv = fmaf(ap, cv, partial[(bb * NCHUNK + c) * D_DIM + d]);
    }
}

__global__ void scan_apply(const float* __restrict__ b,
                           const float* __restrict__ a_sig,
                           const float* __restrict__ carry,
                           __bf16* __restrict__ y) {
    const int gid  = blockIdx.x * blockDim.x + threadIdx.x; // B*NCHUNK*D threads
    const int d    = gid % D_DIM;
    const int rest = gid / D_DIM;
    const int c    = rest % NCHUNK;
    const int bb   = rest / NCHUNK;
    const float av = a_sig[d];
    const int base = (bb * T_LEN + c * CHUNK) * D_DIM + d;
    const float* p = b + base;
    __bf16*      q = y + base;
    float h = carry[(bb * NCHUNK + c) * D_DIM + d];
    for (int t = 0; t < CHUNK; ++t) {
        h = fmaf(av, h, p[t * D_DIM]);
        q[t * D_DIM] = (__bf16)h;
    }
}

// ---------------------------------------------------------------------------
// Launch
// ---------------------------------------------------------------------------
extern "C" void kernel_launch(void* const* d_in, const int* in_sizes, int n_in,
                              void* d_out, int out_size, void* d_ws, size_t ws_size,
                              hipStream_t stream) {
    const float* x   = (const float*)d_in[0];
    const float* W_B = (const float*)d_in[1];
    const float* W_C = (const float*)d_in[2];
    const float* A   = (const float*)d_in[3];
    float* out = (float*)d_out;

    char* ws = (char*)d_ws;
    size_t off = 0;
    __bf16* x_bf  = (__bf16*)(ws + off); off += (size_t)M_DIM * D_DIM * sizeof(__bf16);
    __bf16* wb_bf = (__bf16*)(ws + off); off += (size_t)D_DIM * D_DIM * sizeof(__bf16);
    __bf16* wc_bf = (__bf16*)(ws + off); off += (size_t)D_DIM * D_DIM * sizeof(__bf16);
    float* a_sig  = (float*)(ws + off);  off += D_DIM * sizeof(float);
    float* a_pow  = (float*)(ws + off);  off += D_DIM * sizeof(float);
    float* bbuf   = (float*)(ws + off);  off += (size_t)M_DIM * D_DIM * sizeof(float);
    float* part   = (float*)(ws + off);  off += (size_t)B_SZ * NCHUNK * D_DIM * sizeof(float);
    float* carr   = (float*)(ws + off);  off += (size_t)B_SZ * NCHUNK * D_DIM * sizeof(float);
    __bf16* y_bf  = (__bf16*)(ws + off); off += (size_t)M_DIM * D_DIM * sizeof(__bf16);

    const int n_x = M_DIM * D_DIM;       // 16,777,216
    const int n_w = D_DIM * D_DIM;       // 262,144

    cvt_f32_to_bf16<<<(n_x + 255) / 256, 256, 0, stream>>>(x, x_bf, n_x);
    cvt_f32_to_bf16<<<(n_w + 255) / 256, 256, 0, stream>>>(W_B, wb_bf, n_w);
    cvt_f32_to_bf16<<<(n_w + 255) / 256, 256, 0, stream>>>(W_C, wc_bf, n_w);
    prep_decay<<<(D_DIM + 255) / 256, 256, 0, stream>>>(A, a_sig, a_pow);

    dim3 ggrid(D_DIM / TN, M_DIM / TM);  // (8, 128)
    gemm_bf16_wmma<<<ggrid, 256, 0, stream>>>(x_bf, wb_bf, bbuf);

    const int n_seq_chunks = B_SZ * NCHUNK * D_DIM;   // 131,072
    scan_partial<<<n_seq_chunks / 256, 256, 0, stream>>>(bbuf, a_sig, part);
    scan_carry<<<(B_SZ * D_DIM) / 256, 256, 0, stream>>>(part, a_pow, carr);
    scan_apply<<<n_seq_chunks / 256, 256, 0, stream>>>(bbuf, a_sig, carr, y_bf);

    gemm_bf16_wmma<<<ggrid, 256, 0, stream>>>(y_bf, wc_bf, out);
}
